// BarkSelfAttention_55327768707810
// MI455X (gfx1250) — matrix operand order repaired and checked by reading, not compiled, not run on hardware
//
#include <hip/hip_runtime.h>

typedef _Float16 v16h __attribute__((ext_vector_type(16)));
typedef float    v8f  __attribute__((ext_vector_type(8)));
typedef unsigned int u32x4 __attribute__((ext_vector_type(4)));
typedef int          i32x8 __attribute__((ext_vector_type(8)));
typedef int          i32x4 __attribute__((ext_vector_type(4)));

union AFrag { v16h v; _Float16 e[16]; };
union Acc   { v8f  v; float    e[8];  };

constexpr int BB = 4;          // batch
constexpr int S  = 1024;       // sequence
constexpr int D  = 1024;       // model dim
constexpr int H  = 16;         // heads
constexpr int HD = 64;         // head dim
constexpr int N3 = 3 * D;      // qkv width
constexpr int PART = BB * H * S * HD;   // elements per q/k/v buffer

// A-fragment K index for 16-bit 16x32 A matrix (contiguous order):
// lanes 0-15 cover K {0..15}, lanes 16-31 cover K {16..31}
static __device__ __forceinline__ int kpair(int v, int hi) {
    return 2 * v + 16 * hi;
}

// Generic LDS pointer -> DS byte offset (ISA 10.2: LDS aperture address is
// the low 32 bits of the flat address).
static __device__ __forceinline__ unsigned lds_off32(const void* p) {
    return (unsigned)(unsigned long long)p;
}

// Issue a TDM 2-D tile load (global -> LDS).  D# per ISA ch.8:
//  group0: count=1 | lds_addr | global_addr[31:0] | global_addr[56:32], type=2
//  group1: data_size | tensor_dim0/1 (huge, never OOB) | tile_dim0/1 | stride0
//  remaining groups: zero (2-D tile, iterate/gather off)
// amdgpu-toolchain (clang-23) form: 6 args, trailing i32x8 group + cpol.
static __device__ __forceinline__ void tdm_load_2d(
        unsigned lds_byte_off, const void* gptr, unsigned data_size_log2,
        unsigned tile_x, unsigned tile_y, unsigned stride_x) {
    const unsigned long long ga = (unsigned long long)gptr;
    u32x4 g0;
    g0[0] = 1u;                                   // count = 1 valid descriptor
    g0[1] = lds_byte_off;                         // lds_addr
    g0[2] = (unsigned)ga;                         // global_addr[31:0]
    g0[3] = ((unsigned)(ga >> 32) & 0x01FFFFFFu)  // global_addr[56:32]
            | 0x80000000u;                        // type = 2 ("image")
    i32x8 g1;
    g1[0] = (int)(data_size_log2 << 16);          // data_size; mask/flags = 0
    g1[1] = 0;                                    // tensor_dim0[15:0] << 16 (=0)
    g1[2] = 0x4000;                               // tensor_dim0 = 1<<30, dim1 lo
    g1[3] = (int)(0x4000u | (tile_x << 16));      // tensor_dim1 = 1<<30 | tile_dim0
    g1[4] = (int)tile_y;                          // tile_dim1 | tile_dim2 = 0
    g1[5] = (int)stride_x;                        // tensor_dim0_stride[31:0]
    g1[6] = 0;                                    // stride0 hi | stride1 lo
    g1[7] = 0;
    const i32x4 gz4 = {0, 0, 0, 0};
    const i32x8 gz8 = {0, 0, 0, 0, 0, 0, 0, 0};
    __builtin_amdgcn_tensor_load_to_lds(g0, g1, gz4, gz4, gz8, 0);
}

// ---------------------------------------------------------------------------
// Kernel 1: QKV projection.  C[4096,3072] = A[4096,1024] @ W[1024,3072] + b
// 128x128 block tile, TDM-staged operands, 8 waves x (2x4) WMMA tiles.
// Output scattered as f16 into q/k/v buffers laid out [B][H][S][HD].
// ---------------------------------------------------------------------------
__global__ __launch_bounds__(256) void qkv_gemm(
        const float* __restrict__ A, const float* __restrict__ W,
        const float* __restrict__ bias, _Float16* __restrict__ qkv) {
    __shared__ float As32[128][32];   // A tile, row-major [m][k]
    __shared__ float Bs32[32][128];   // W tile, row-major [k][n]

    const int t = threadIdx.x;
    const int lane = t & 31, wave = t >> 5;
    const int ln = lane & 15, hi = lane >> 4;
    const int m0 = blockIdx.y * 128;
    const int n0 = blockIdx.x * 128;
    const int wm0 = (wave >> 1) * 32, wn0 = (wave & 1) * 64;

    Acc acc[2][4];
    #pragma unroll
    for (int i = 0; i < 2; i++)
        #pragma unroll
        for (int j = 0; j < 4; j++)
            #pragma unroll
            for (int v = 0; v < 8; v++) acc[i][j].e[v] = 0.f;

    for (int k0 = 0; k0 < D; k0 += 32) {
        __syncthreads();                       // previous tile fully consumed
        if (wave == 0) {
            tdm_load_2d(lds_off32(&As32[0][0]),
                        A + (size_t)m0 * D + k0, 2u, 32u, 128u, (unsigned)D);
            tdm_load_2d(lds_off32(&Bs32[0][0]),
                        W + (size_t)k0 * N3 + n0, 2u, 128u, 32u, (unsigned)N3);
            __builtin_amdgcn_s_wait_tensorcnt((short)0);
        }
        __syncthreads();                       // tiles visible to all waves

        AFrag af[2], bf[4];
        #pragma unroll
        for (int mi = 0; mi < 2; mi++) {
            const int m = wm0 + mi * 16 + ln;
            #pragma unroll
            for (int v = 0; v < 8; v++) {
                const int kk = kpair(v, hi);
                af[mi].e[2 * v]     = (_Float16)As32[m][kk];
                af[mi].e[2 * v + 1] = (_Float16)As32[m][kk + 1];
            }
        }
        #pragma unroll
        for (int ni = 0; ni < 4; ni++) {
            const int n = wn0 + ni * 16 + ln;
            #pragma unroll
            for (int i = 0; i < 16; i++)
                bf[ni].e[i] = (_Float16)Bs32[hi * 16 + i][n];
        }
        #pragma unroll
        for (int mi = 0; mi < 2; mi++)
            #pragma unroll
            for (int ni = 0; ni < 4; ni++)
                acc[mi][ni].v = __builtin_amdgcn_wmma_f32_16x16x32_f16(
                    false, af[mi].v, false, bf[ni].v, (short)0, acc[mi][ni].v,
                    false, false);
    }

    #pragma unroll
    for (int mi = 0; mi < 2; mi++)
        #pragma unroll
        for (int ni = 0; ni < 4; ni++) {
            const int gn = n0 + wn0 + ni * 16 + ln;
            const int part = gn >> 10, rem = gn & 1023;
            const int h = rem >> 6, d = rem & 63;
            const float bv = bias[gn];
            #pragma unroll
            for (int v = 0; v < 8; v++) {
                const int gm = m0 + wm0 + mi * 16 + v + 8 * hi;
                const int b = gm >> 10, s = gm & 1023;
                const size_t idx = (size_t)part * PART +
                                   (((size_t)b * H + h) * S + s) * HD + d;
                qkv[idx] = (_Float16)(acc[mi][ni].e[v] + bv);
            }
        }
}

// ---------------------------------------------------------------------------
// Kernel 2: causal flash attention.  4 waves/block; wave owns 16 query rows
// of one (b,h).  Online softmax kept entirely in WMMA C/D register layout.
// ---------------------------------------------------------------------------
__global__ __launch_bounds__(128) void attn_fwd(
        const _Float16* __restrict__ Q, const _Float16* __restrict__ K,
        const _Float16* __restrict__ V, _Float16* __restrict__ ctx) {
    __shared__ _Float16 Pl[4][16][36];   // per-wave P tile (16x32 f16, padded)

    const int t = threadIdx.x, lane = t & 31, wave = t >> 5;
    const int ln = lane & 15, hi = lane >> 4;
    const int qblk = blockIdx.x, bh = blockIdx.y;
    const int b = bh >> 4, h = bh & 15;
    const int qw = qblk * 64 + wave * 16;

    const _Float16* qp = Q + (size_t)bh * S * HD;
    const _Float16* kp = K + (size_t)bh * S * HD;
    const _Float16* vp = V + (size_t)bh * S * HD;

    AFrag qf[2];
    #pragma unroll
    for (int dk = 0; dk < 2; dk++) {
        const int m = qw + ln;
        #pragma unroll
        for (int v = 0; v < 8; v++) {
            const int kk = kpair(v, hi) + dk * 32;
            qf[dk].e[2 * v]     = qp[(size_t)m * HD + kk];
            qf[dk].e[2 * v + 1] = qp[(size_t)m * HD + kk + 1];
        }
    }

    float mrow[8], lrow[8];
    Acc cacc[4];
    #pragma unroll
    for (int v = 0; v < 8; v++) { mrow[v] = -1e30f; lrow[v] = 0.f; }
    #pragma unroll
    for (int vt = 0; vt < 4; vt++)
        #pragma unroll
        for (int v = 0; v < 8; v++) cacc[vt].e[v] = 0.f;

    const int nchunk = (qw + 16 + 31) >> 5;
    for (int c = 0; c < nchunk; ++c) {
        const int kb = c * 32;
        float sc[2][8];
        #pragma unroll
        for (int tt = 0; tt < 2; tt++) {
            const int key = kb + tt * 16 + ln;
            AFrag kf0, kf1;
            #pragma unroll
            for (int i = 0; i < 16; i++) {
                kf0.e[i] = kp[(size_t)key * HD + hi * 16 + i];
                kf1.e[i] = kp[(size_t)key * HD + 32 + hi * 16 + i];
            }
            Acc s;
            #pragma unroll
            for (int v = 0; v < 8; v++) s.e[v] = 0.f;
            s.v = __builtin_amdgcn_wmma_f32_16x16x32_f16(
                false, qf[0].v, false, kf0.v, (short)0, s.v, false, false);
            s.v = __builtin_amdgcn_wmma_f32_16x16x32_f16(
                false, qf[1].v, false, kf1.v, (short)0, s.v, false, false);
            #pragma unroll
            for (int v = 0; v < 8; v++) {
                const int qrow = qw + v + 8 * hi;
                const float val = s.e[v] * 0.125f;      // 1/sqrt(64)
                sc[tt][v] = (key <= qrow) ? val : -1e30f;
            }
        }
        float scale[8];
        #pragma unroll
        for (int v = 0; v < 8; v++) {
            float mx = fmaxf(sc[0][v], sc[1][v]);
            mx = fmaxf(mx, __shfl_xor(mx, 1, 32));
            mx = fmaxf(mx, __shfl_xor(mx, 2, 32));
            mx = fmaxf(mx, __shfl_xor(mx, 4, 32));
            mx = fmaxf(mx, __shfl_xor(mx, 8, 32));
            const float mnew = fmaxf(mrow[v], mx);
            const float sca = __expf(mrow[v] - mnew);
            mrow[v] = mnew; scale[v] = sca;
            const float p0 = __expf(sc[0][v] - mnew);
            const float p1 = __expf(sc[1][v] - mnew);
            sc[0][v] = p0; sc[1][v] = p1;
            float rs = p0 + p1;
            rs += __shfl_xor(rs, 1, 32);
            rs += __shfl_xor(rs, 2, 32);
            rs += __shfl_xor(rs, 4, 32);
            rs += __shfl_xor(rs, 8, 32);
            lrow[v] = lrow[v] * sca + rs;
        }
        #pragma unroll
        for (int vt = 0; vt < 4; vt++)
            #pragma unroll
            for (int v = 0; v < 8; v++) cacc[vt].e[v] *= scale[v];

        #pragma unroll
        for (int tt = 0; tt < 2; tt++)
            #pragma unroll
            for (int v = 0; v < 8; v++)
                Pl[wave][v + 8 * hi][tt * 16 + ln] = (_Float16)sc[tt][v];
        __asm__ volatile("s_wait_dscnt 0" ::: "memory");

        AFrag pf;
        #pragma unroll
        for (int v = 0; v < 8; v++) {
            const int kk = kpair(v, hi);
            pf.e[2 * v]     = Pl[wave][ln][kk];
            pf.e[2 * v + 1] = Pl[wave][ln][kk + 1];
        }
        #pragma unroll
        for (int vt = 0; vt < 4; vt++) {
            AFrag vf;
            #pragma unroll
            for (int i = 0; i < 16; i++)
                vf.e[i] = vp[(size_t)(kb + hi * 16 + i) * HD + vt * 16 + ln];
            cacc[vt].v = __builtin_amdgcn_wmma_f32_16x16x32_f16(
                false, pf.v, false, vf.v, (short)0, cacc[vt].v, false, false);
        }
    }

    #pragma unroll
    for (int v = 0; v < 8; v++) {
        const int s = qw + v + 8 * hi;
        const float inv = 1.0f / lrow[v];
        #pragma unroll
        for (int vt = 0; vt < 4; vt++) {
            const int col = h * HD + vt * 16 + ln;
            ctx[((size_t)b * S + s) * D + col] = (_Float16)(cacc[vt].e[v] * inv);
        }
    }
}

// ---------------------------------------------------------------------------
// Kernel 3: output projection.  out[4096,1024] = ctx_f16 @ W[1024,1024] + b
// TDM-staged: A tile is f16 (data_size=2B), W tile f32.
// ---------------------------------------------------------------------------
__global__ __launch_bounds__(256) void out_gemm(
        const _Float16* __restrict__ A, const float* __restrict__ W,
        const float* __restrict__ bias, float* __restrict__ out) {
    __shared__ _Float16 As16[128][32];
    __shared__ float    Bs32[32][128];

    const int t = threadIdx.x;
    const int lane = t & 31, wave = t >> 5;
    const int ln = lane & 15, hi = lane >> 4;
    const int m0 = blockIdx.y * 128;
    const int n0 = blockIdx.x * 128;
    const int wm0 = (wave >> 1) * 32, wn0 = (wave & 1) * 64;

    Acc acc[2][4];
    #pragma unroll
    for (int i = 0; i < 2; i++)
        #pragma unroll
        for (int j = 0; j < 4; j++)
            #pragma unroll
            for (int v = 0; v < 8; v++) acc[i][j].e[v] = 0.f;

    for (int k0 = 0; k0 < D; k0 += 32) {
        __syncthreads();
        if (wave == 0) {
            tdm_load_2d(lds_off32(&As16[0][0]),
                        A + (size_t)m0 * D + k0, 1u, 32u, 128u, (unsigned)D);
            tdm_load_2d(lds_off32(&Bs32[0][0]),
                        W + (size_t)k0 * D + n0, 2u, 128u, 32u, (unsigned)D);
            __builtin_amdgcn_s_wait_tensorcnt((short)0);
        }
        __syncthreads();

        AFrag af[2], bf[4];
        #pragma unroll
        for (int mi = 0; mi < 2; mi++) {
            const int m = wm0 + mi * 16 + ln;
            #pragma unroll
            for (int v = 0; v < 8; v++) {
                const int kk = kpair(v, hi);
                af[mi].e[2 * v]     = As16[m][kk];
                af[mi].e[2 * v + 1] = As16[m][kk + 1];
            }
        }
        #pragma unroll
        for (int ni = 0; ni < 4; ni++) {
            const int n = wn0 + ni * 16 + ln;
            #pragma unroll
            for (int i = 0; i < 16; i++)
                bf[ni].e[i] = (_Float16)Bs32[hi * 16 + i][n];
        }
        #pragma unroll
        for (int mi = 0; mi < 2; mi++)
            #pragma unroll
            for (int ni = 0; ni < 4; ni++)
                acc[mi][ni].v = __builtin_amdgcn_wmma_f32_16x16x32_f16(
                    false, af[mi].v, false, bf[ni].v, (short)0, acc[mi][ni].v,
                    false, false);
    }

    #pragma unroll
    for (int mi = 0; mi < 2; mi++)
        #pragma unroll
        for (int ni = 0; ni < 4; ni++) {
            const int gn = n0 + wn0 + ni * 16 + ln;
            const float bv = bias[gn];
            #pragma unroll
            for (int v = 0; v < 8; v++) {
                const int gm = m0 + wm0 + mi * 16 + v + 8 * hi;
                out[(size_t)gm * D + gn] = acc[mi][ni].e[v] + bv;
            }
        }
}

// ---------------------------------------------------------------------------
extern "C" void kernel_launch(void* const* d_in, const int* in_sizes, int n_in,
                              void* d_out, int out_size, void* d_ws,
                              size_t ws_size, hipStream_t stream) {
    (void)in_sizes; (void)n_in; (void)out_size; (void)ws_size;
    const float* hs    = (const float*)d_in[0];
    const float* att_w = (const float*)d_in[1];
    const float* att_b = (const float*)d_in[2];
    const float* out_w = (const float*)d_in[3];
    const float* out_b = (const float*)d_in[4];
    float* out = (float*)d_out;

    _Float16* ws   = (_Float16*)d_ws;
    _Float16* qbuf = ws;                       // [B][H][S][HD] f16
    _Float16* kbuf = ws + (size_t)PART;
    _Float16* vbuf = ws + (size_t)2 * PART;
    _Float16* ctxb = ws + (size_t)3 * PART;    // [B][S][D]     f16

    qkv_gemm<<<dim3(N3 / 128, (BB * S) / 128), 256, 0, stream>>>(hs, att_w, att_b, ws);
    attn_fwd<<<dim3(S / 64, BB * H), 128, 0, stream>>>(qbuf, kbuf, vbuf, ctxb);
    out_gemm<<<dim3(D / 128, (BB * S) / 128), 256, 0, stream>>>(ctxb, out_w, out_b, out);
}
